// TSPRGCNActionNet_47931835023898
// MI455X (gfx1250) — compile-verified, matrix operand order checked by reading, not run on hardware
//
#include <hip/hip_runtime.h>
#include <hip/hip_bf16.h>

// ---------------------------------------------------------------------------
// Problem constants (from reference): B=8, V=200, H=64, NLAYERS=3, VOC=3
// ---------------------------------------------------------------------------
#define B_  8
#define V_  200
#define H_  64
#define P_  19900                 // C(200,2)
#define ME_ (B_*V_*V_)            // 320000 edge rows (multiple of 16)
#define MX_ (B_*V_)               // 1600 node rows   (multiple of 16)
#define MQ_ (B_*P_)               // 159200 pair rows (multiple of 16)

typedef __bf16 bf16;
typedef __attribute__((ext_vector_type(16))) __bf16 v16bf;
typedef __attribute__((ext_vector_type(8)))  __bf16 v8bf;
typedef __attribute__((ext_vector_type(8)))  float  v8f;

// ---------------------------------------------------------------------------
// Pack a row-major f32 weight matrix W[K][64] into per-lane WMMA B-fragments.
// Fragment layout (16-bit B, 32x16 tile, wave32):
//   lane l: column n = nt*16 + (l&15), kbase = (l>>4)*8
//   element i<8  -> k = kt*32 + kbase + i
//   element i>=8 -> k = kt*32 + kbase + i + 8
// Packed index: ((kt*4 + nt)*32 + lane)*16 + i   (N fixed = 64 -> 4 n-tiles)
// ---------------------------------------------------------------------------
__global__ void k_pack_w(const float* __restrict__ W, bf16* __restrict__ out, int K) {
    int idx = blockIdx.x * blockDim.x + threadIdx.x;
    if (idx >= K * 64) return;
    int i    = idx & 15;
    int lane = (idx >> 4) & 31;
    int slot = idx >> 9;            // kt*4 + nt
    int nt = slot & 3;
    int kt = slot >> 2;
    int n  = nt * 16 + (lane & 15);
    int kk = kt * 32 + ((lane >> 4) << 3) + (i < 8 ? i : i + 8);
    out[idx] = (bf16)W[kk * 64 + n];
}

// ---------------------------------------------------------------------------
// Epilogue (N = 64), specialized at compile time:
//  MODE 0: out = acc+bias
//  MODE 1: out = relu(acc+bias)
//  MODE 2: edge fuse: val = acc+bias+Vx[b,u,:]+Vx[b,v,:]; outf=e_tmp; gate=sigmoid
// C/D layout: vgpr i, lanes 0-15 -> M=i, lanes 16-31 -> M=8+i; n = nt*16+(lane&15)
// ---------------------------------------------------------------------------
template<int MODE, bool WB>
__device__ __forceinline__ void gemm_epilogue(const v8f* acc, int mtile, int lane,
                                              const float* __restrict__ bias,
                                              float* __restrict__ outf, bf16* __restrict__ outb,
                                              const float* __restrict__ vx,
                                              float* __restrict__ gate) {
    int roff  = (lane >> 4) << 3;          // 0 or 8
    int mbase = mtile * 16 + roff;
    int nlo   = lane & 15;

    // tile-uniform batch index for the edge-fused path (40000 % 16 == 0:
    // a 16-row tile never straddles a batch)
    int rembase = 0;
    const float* vxb = nullptr;
    if (MODE == 2) {
        int bb  = (mtile * 16) / (V_ * V_);
        rembase = mtile * 16 - bb * (V_ * V_);
        vxb     = vx + (size_t)bb * V_ * 64;
    }
#pragma unroll
    for (int nt = 0; nt < 4; ++nt) {
        int n = nt * 16 + nlo;
        float bsv = bias[n];
#pragma unroll
        for (int i = 0; i < 8; ++i) {
            int m = mbase + i;
            float val = acc[nt][i] + bsv;
            size_t o = (size_t)m * 64 + n;
            if (MODE == 2) {
                int r  = rembase + roff + i;     // < 40000
                int u  = r / V_;
                int vv = r - u * V_;
                val += vxb[(size_t)u * 64 + n] + vxb[(size_t)vv * 64 + n];
                outf[o] = val;                               // e_tmp
                gate[o] = 1.f / (1.f + __expf(-val));        // sigmoid gate
            } else {
                if (MODE == 1) val = fmaxf(val, 0.f);
                outf[o] = val;
                if (WB) outb[o] = (bf16)val;
            }
        }
    }
}

// ---------------------------------------------------------------------------
// K=64 fast path: the whole packed weight (8 fragments = 64 VGPRs/lane) is
// loaded ONCE per wave and held in registers; each wave then streams
// `tilesPerWave` consecutive 16-row M-tiles: 4 x b128 A-loads -> 8 WMMAs.
// __launch_bounds__(256,2): 8-wave blocks = 2 waves/SIMD floor, raises the
// VGPR cap so the B fragments stay register-resident (no scratch spills).
// ---------------------------------------------------------------------------
template<int MODE, bool WB>
__global__ void __launch_bounds__(256, 2)
k_wmma_gemm64(const bf16* __restrict__ A, const bf16* __restrict__ Bp,
              const float* __restrict__ bias,
              float* __restrict__ outf, bf16* __restrict__ outb,
              int M, const float* __restrict__ vx, float* __restrict__ gate,
              int tilesPerWave) {
    int lane = threadIdx.x & 31;
    int wave = blockIdx.x * (blockDim.x >> 5) + (threadIdx.x >> 5);
    int kb   = (lane >> 4) << 3;                 // 0 or 8

    // B fragments resident in registers for the whole kernel
    v16bf bfr[8];
#pragma unroll
    for (int f = 0; f < 8; ++f)
        bfr[f] = *(const v16bf*)(Bp + ((size_t)f * 32 + lane) * 16);

    for (int t = 0; t < tilesPerWave; ++t) {
        int mtile = wave * tilesPerWave + t;
        if (mtile * 16 >= M) return;             // uniform per wave: EXEC stays all-1
        int row = mtile * 16 + (lane & 15);
        const bf16* arow = A + (size_t)row * 64;
        union { v16bf v; v8bf h[2]; } a0, a1;
        a0.h[0] = *(const v8bf*)(arow + kb);
        a0.h[1] = *(const v8bf*)(arow + kb + 16);
        a1.h[0] = *(const v8bf*)(arow + 32 + kb);
        a1.h[1] = *(const v8bf*)(arow + 32 + kb + 16);
        if (t + 1 < tilesPerWave)
            __builtin_prefetch(arow + 16 * 64, 0, 0);   // next tile's A rows

        v8f acc[4] = {};
#pragma unroll
        for (int nt = 0; nt < 4; ++nt) {
            acc[nt] = __builtin_amdgcn_wmma_f32_16x16x32_bf16(
                false, a0.v, false, bfr[nt], (short)0, acc[nt], false, false);
            acc[nt] = __builtin_amdgcn_wmma_f32_16x16x32_bf16(
                false, a1.v, false, bfr[4 + nt], (short)0, acc[nt], false, false);
        }
        gemm_epilogue<MODE, WB>(acc, mtile, lane, bias, outf, outb, vx, gate);
    }
}

// ---------------------------------------------------------------------------
// Generic-K path (used for K=256 pre-MLP, MODE=1 + bf16 out): packed weight
// staged in LDS once per block; B fragments read via ds_load_b128.
// ---------------------------------------------------------------------------
__global__ void __launch_bounds__(256, 2)
k_wmma_gemm_lds(const bf16* __restrict__ A, const bf16* __restrict__ Bp,
                const float* __restrict__ bias,
                float* __restrict__ outf, bf16* __restrict__ outb,
                int M, int K, int tilesPerWave) {
    extern __shared__ bf16 smem[];               // K*64 bf16 packed fragments
    int nvec = (K * 64) >> 3;                    // v8bf chunks
    for (int tix = threadIdx.x; tix < nvec; tix += blockDim.x)
        ((v8bf*)smem)[tix] = ((const v8bf*)Bp)[tix];
    __syncthreads();

    int lane = threadIdx.x & 31;
    int wave = blockIdx.x * (blockDim.x >> 5) + (threadIdx.x >> 5);
    int kb   = (lane >> 4) << 3;
    int ktiles = K >> 5;

    for (int t = 0; t < tilesPerWave; ++t) {
        int mtile = wave * tilesPerWave + t;
        if (mtile * 16 >= M) return;             // uniform per wave
        int row = mtile * 16 + (lane & 15);
        const bf16* arow = A + (size_t)row * K;

        v8f acc[4] = {};
        for (int kt = 0; kt < ktiles; ++kt) {
            union { v16bf v; v8bf h[2]; } a;
            a.h[0] = *(const v8bf*)(arow + kt * 32 + kb);
            a.h[1] = *(const v8bf*)(arow + kt * 32 + kb + 16);
            const bf16* bp = smem + ((size_t)kt * 4 * 32 + lane) * 16;
#pragma unroll
            for (int nt = 0; nt < 4; ++nt) {
                v16bf b = *(const v16bf*)(bp + (size_t)nt * 32 * 16);
                acc[nt] = __builtin_amdgcn_wmma_f32_16x16x32_bf16(
                    false, a.v, false, b, (short)0, acc[nt], false, false);
            }
        }
        gemm_epilogue<1, true>(acc, mtile, lane, bias, outf, outb, nullptr, nullptr);
    }
}

// x[b,v,h] = coord[b,v,:] @ node_w   (NODE_DIM=2)
__global__ void k_node_embed(const float* __restrict__ coord, const float* __restrict__ nw,
                             float* __restrict__ xf, bf16* __restrict__ xb) {
    int idx = blockIdx.x * blockDim.x + threadIdx.x;
    if (idx >= MX_ * 64) return;
    int h = idx & 63; int m = idx >> 6;
    float s = coord[m * 2] * nw[h] + coord[m * 2 + 1] * nw[64 + h];
    xf[idx] = s; xb[idx] = (bf16)s;
}

// e = concat(vals*val_w, emb0[tour]+emb1[best])
__global__ void k_build_e(const float* __restrict__ vals, const int* __restrict__ tour,
                          const int* __restrict__ best, const float* __restrict__ val_w,
                          const float* __restrict__ emb0, const float* __restrict__ emb1,
                          float* __restrict__ ef, bf16* __restrict__ eb) {
    size_t idx = (size_t)blockIdx.x * blockDim.x + threadIdx.x;
    if (idx >= (size_t)ME_ * 64) return;
    int h = (int)(idx & 63); size_t m = idx >> 6;
    float r;
    if (h < 32) r = vals[m] * val_w[h];
    else        r = emb0[tour[m] * 32 + (h - 32)] + emb1[best[m] * 32 + (h - 32)];
    ef[idx] = r; eb[idx] = (bf16)r;
}

// agg[b,u,h] = sum_v gate*Vx2[b,v,h]; gsum = sum_v gate
__global__ void k_agg(const float* __restrict__ gate, const float* __restrict__ vx2,
                      float* __restrict__ agg, float* __restrict__ gsum) {
    int idx = blockIdx.x * blockDim.x + threadIdx.x;
    if (idx >= MX_ * 64) return;
    int h = idx & 63; int bu = idx >> 6; int b = bu / V_;
    const float* g  = gate + ((size_t)bu * V_) * 64 + h;
    const float* v2 = vx2  + ((size_t)b * V_) * 64 + h;
    float s = 0.f, a = 0.f;
    for (int v = 0; v < V_; ++v) {
        float gv = g[(size_t)v * 64];
        s += gv; a += gv * v2[(size_t)v * 64];
    }
    agg[idx] = a; gsum[idx] = s;
}

__global__ void k_xtmp(const float* __restrict__ ux, const float* __restrict__ agg,
                       const float* __restrict__ gsum, float* __restrict__ xt) {
    int idx = blockIdx.x * blockDim.x + threadIdx.x;
    if (idx >= MX_ * 64) return;
    xt[idx] = ux[idx] + agg[idx] / (1e-20f + gsum[idx]);
}

// per-feature batch mean/var (training-mode BN stats): one block per feature
__global__ void k_stats(const float* __restrict__ t, float* __restrict__ mean,
                        float* __restrict__ var, int M) {
    int f = blockIdx.x, tid = threadIdx.x;
    float s = 0.f, s2 = 0.f;
    for (int m = tid; m < M; m += blockDim.x) {
        float v = t[(size_t)m * 64 + f];
        s += v; s2 += v * v;
    }
    __shared__ float sh[256], sh2[256];
    sh[tid] = s; sh2[tid] = s2; __syncthreads();
    for (int st = 128; st > 0; st >>= 1) {
        if (tid < st) { sh[tid] += sh[tid + st]; sh2[tid] += sh2[tid + st]; }
        __syncthreads();
    }
    if (tid == 0) {
        float mu = sh[0] / M;
        mean[f] = mu; var[f] = sh2[0] / M - mu * mu;
    }
}

// base += relu(bn(tmp)); also refresh bf16 mirror
__global__ void k_bnres(float* __restrict__ base, const float* __restrict__ tmp,
                        const float* __restrict__ gamma, const float* __restrict__ beta,
                        const float* __restrict__ mean, const float* __restrict__ var,
                        bf16* __restrict__ bfm, size_t total) {
    size_t idx = (size_t)blockIdx.x * blockDim.x + threadIdx.x;
    if (idx >= total) return;
    int h = (int)(idx & 63);
    float nb = gamma[h] * (tmp[idx] - mean[h]) * rsqrtf(var[h] + 1e-5f) + beta[h];
    float r  = base[idx] + fmaxf(nb, 0.f);
    base[idx] = r;
    if (bfm) bfm[idx] = (bf16)r;
}

// per-batch tour-edge list (lexicographic u<v) + global triu pair index tables
__global__ void k_edges(const int* __restrict__ x_tour, int* __restrict__ el,
                        int* __restrict__ rs, int* __restrict__ cs) {
    int b = threadIdx.x;
    if (b < B_) {
        int cnt = 0;
        for (int u = 0; u < V_; ++u)
            for (int v = u + 1; v < V_; ++v)
                if (x_tour[((size_t)b * V_ + u) * V_ + v] > 0 && cnt < V_) {
                    el[(b * V_ + cnt) * 2] = u; el[(b * V_ + cnt) * 2 + 1] = v; ++cnt;
                }
    }
    if (threadIdx.x == 0) {
        int p = 0;
        for (int r = 0; r < V_; ++r)
            for (int c = r + 1; c < V_; ++c) { rs[p] = r; cs[p] = c; ++p; }
    }
}

// quad[b,p,0:256] = concat(e[b,us,vs], e[b,xs,ys], e[b,us,xs], e[b,vs,ys]) -> bf16
// meta[b,p] = (u,v,x,y) un-normalized (for the returned edge indices)
__global__ void k_quad(const float* __restrict__ e, const int* __restrict__ el,
                       const int* __restrict__ rs, const int* __restrict__ cs,
                       const unsigned char* __restrict__ dir,
                       int* __restrict__ meta, bf16* __restrict__ quad) {
    int bp = blockIdx.x;
    int b = bp / P_, p = bp - b * P_;
    int r = rs[p], c = cs[p];
    int u = el[(b * V_ + r) * 2], v = el[(b * V_ + r) * 2 + 1];
    int x = el[(b * V_ + c) * 2], y = el[(b * V_ + c) * 2 + 1];
    int us = u, vs = v;
    if (!dir[((size_t)b * V_ + u) * V_ + v]) { us = v; vs = u; }
    int xs = x, ys = y;
    if (!dir[((size_t)b * V_ + x) * V_ + y]) { xs = y; ys = x; }
    int j = threadIdx.x;            // 0..255
    int seg = j >> 6, h = j & 63;
    int a0, a1;
    if      (seg == 0) { a0 = us; a1 = vs; }
    else if (seg == 1) { a0 = xs; a1 = ys; }
    else if (seg == 2) { a0 = us; a1 = xs; }
    else               { a0 = vs; a1 = ys; }
    float val = e[(((size_t)b * V_ + a0) * V_ + a1) * 64 + h];
    quad[(size_t)bp * 256 + j] = (bf16)val;
    if (j == 0) { int* m = meta + (size_t)bp * 4; m[0] = u; m[1] = v; m[2] = x; m[3] = y; }
}

// final act projection 64 -> 1
__global__ void k_dotout(const float* __restrict__ hid, const float* __restrict__ w,
                         const float* __restrict__ bias, float* __restrict__ logits, int M) {
    int m = blockIdx.x * blockDim.x + threadIdx.x;
    if (m >= M) return;
    float s = bias[0];
    for (int h = 0; h < 64; ++h) s += hid[(size_t)m * 64 + h] * w[h];
    logits[m] = s;
}

// deterministic action selection + log_softmax(logits)[b, a]; one block per batch
__global__ void k_sample(const float* __restrict__ logits, const int* __restrict__ meta,
                         float* __restrict__ out) {
    int b = blockIdx.x, tid = threadIdx.x;
    __shared__ float smax[256]; __shared__ int sidx[256]; __shared__ float ssum[256];
    float mx = -1e30f; int mi = 0;
    for (int p = tid; p < P_; p += 256) {
        float l = logits[(size_t)b * P_ + p];
        if (l > mx) { mx = l; mi = p; }
    }
    smax[tid] = mx; sidx[tid] = mi; __syncthreads();
    for (int s = 128; s > 0; s >>= 1) {
        if (tid < s && smax[tid + s] > smax[tid]) { smax[tid] = smax[tid + s]; sidx[tid] = sidx[tid + s]; }
        __syncthreads();
    }
    float m = smax[0]; int a = sidx[0];
    float se = 0.f;
    for (int p = tid; p < P_; p += 256) se += __expf(logits[(size_t)b * P_ + p] - m);
    ssum[tid] = se; __syncthreads();
    for (int s = 128; s > 0; s >>= 1) {
        if (tid < s) ssum[tid] += ssum[tid + s];
        __syncthreads();
    }
    if (tid == 0) {
        float pi = logits[(size_t)b * P_ + a] - m - __logf(ssum[0]);
        const int* mt = meta + ((size_t)b * P_ + a) * 4;
        out[b * 6 + 0] = (float)b; out[b * 6 + 1] = (float)mt[0]; out[b * 6 + 2] = (float)mt[1];
        out[b * 6 + 3] = (float)b; out[b * 6 + 4] = (float)mt[2]; out[b * 6 + 5] = (float)mt[3];
        out[48 + b] = pi;
        out[56 + b] = (float)a;
    }
}

// ---------------------------------------------------------------------------
// Host side
// ---------------------------------------------------------------------------
extern "C" void kernel_launch(void* const* d_in, const int* in_sizes, int n_in,
                              void* d_out, int out_size, void* d_ws, size_t ws_size,
                              hipStream_t stream) {
    (void)in_sizes; (void)n_in; (void)out_size; (void)ws_size;

    // ---- input pointers (setup_inputs dict order; params tree-flattened
    //      with dict keys sorted: act, emb0, emb1, layers, node_w, pre, val_w)
    const float* evals = (const float*)d_in[1];
    const float* coord = (const float*)d_in[2];
    const int*   tour  = (const int*)d_in[3];
    const int*   best  = (const int*)d_in[4];
    const unsigned char* dirb = (const unsigned char*)d_in[5];
    // act: hidden 4x(w,b) = 6..13, out (w,b) = 14,15
    // emb0=16, emb1=17; layers: base 18+12L with
    //   +0 bn_e_g +1 bn_e_b +2 bn_n_g +3 bn_n_b +4 eU_w +5 eU_b +6 eV_w +7 eV_b
    //   +8 nU_w +9 nU_b +10 nV_w +11 nV_b
    // node_w=54; pre: h0_w=55 h0_b=56 out_w=57 out_b=58; val_w=59
    const float* emb0   = (const float*)d_in[16];
    const float* emb1   = (const float*)d_in[17];
    const float* node_w = (const float*)d_in[54];
    const float* val_w  = (const float*)d_in[59];

    // ---- workspace carve-up
    char* w = (char*)d_ws;
    size_t off = 0;
    auto alloc = [&](size_t bytes) -> void* {
        void* p = w + off;
        off = (off + bytes + 255) & ~(size_t)255;
        return p;
    };
    float* e_f    = (float*)alloc((size_t)ME_ * 64 * 4);
    bf16*  e_b    = (bf16*) alloc((size_t)ME_ * 64 * 2);
    float* e_tmp  = (float*)alloc((size_t)ME_ * 64 * 4);
    float* gate   = (float*)alloc((size_t)ME_ * 64 * 4);
    bf16*  quad   = (bf16*) alloc((size_t)MQ_ * 256 * 2);
    float* bufA_f = (float*)alloc((size_t)MQ_ * 64 * 4);
    bf16*  bufA_b = (bf16*) alloc((size_t)MQ_ * 64 * 2);
    float* bufB_f = (float*)alloc((size_t)MQ_ * 64 * 4);
    bf16*  bufB_b = (bf16*) alloc((size_t)MQ_ * 64 * 2);
    float* x_f    = (float*)alloc((size_t)MX_ * 64 * 4);
    bf16*  x_b    = (bf16*) alloc((size_t)MX_ * 64 * 2);
    float* Vx     = (float*)alloc((size_t)MX_ * 64 * 4);
    float* Ux     = (float*)alloc((size_t)MX_ * 64 * 4);
    float* Vx2    = (float*)alloc((size_t)MX_ * 64 * 4);
    float* agg    = (float*)alloc((size_t)MX_ * 64 * 4);
    float* gsum   = (float*)alloc((size_t)MX_ * 64 * 4);
    float* xtmp   = (float*)alloc((size_t)MX_ * 64 * 4);
    float* meanb  = (float*)alloc(64 * 4);
    float* varb   = (float*)alloc(64 * 4);
    float* logits = (float*)alloc((size_t)MQ_ * 4);
    int*   elist  = (int*)  alloc((size_t)B_ * V_ * 2 * 4);
    int*   rs     = (int*)  alloc((size_t)P_ * 4);
    int*   cs     = (int*)  alloc((size_t)P_ * 4);
    int*   meta   = (int*)  alloc((size_t)MQ_ * 4 * 4);
    bf16*  packs  = (bf16*) alloc((size_t)18 * 16384 * 2);   // 18 packed weight slots
    auto pk = [&](int slot) { return packs + (size_t)slot * 16384; };

    // ---- pack all WMMA weight matrices to fragment layout
    // slots: 4L+0=eU_L 4L+1=eV_L 4L+2=nU_L 4L+3=nV_L ; 12=pre_h0 13=pre_out 14..17=act_h0..3
    for (int L = 0; L < 3; ++L) {
        int base = 18 + 12 * L;
        k_pack_w<<<16, 256, 0, stream>>>((const float*)d_in[base + 4],  pk(4 * L + 0), 64);
        k_pack_w<<<16, 256, 0, stream>>>((const float*)d_in[base + 6],  pk(4 * L + 1), 64);
        k_pack_w<<<16, 256, 0, stream>>>((const float*)d_in[base + 8],  pk(4 * L + 2), 64);
        k_pack_w<<<16, 256, 0, stream>>>((const float*)d_in[base + 10], pk(4 * L + 3), 64);
    }
    k_pack_w<<<64, 256, 0, stream>>>((const float*)d_in[55], pk(12), 256);
    k_pack_w<<<16, 256, 0, stream>>>((const float*)d_in[57], pk(13), 64);
    for (int k = 0; k < 4; ++k)
        k_pack_w<<<16, 256, 0, stream>>>((const float*)d_in[6 + 2 * k], pk(14 + k), 64);

    auto nblocks = [](int M, int tpw) { return (M / 16 + 8 * tpw - 1) / (8 * tpw); };

    // ---- initial embeddings
    k_node_embed<<<(MX_ * 64 + 255) / 256, 256, 0, stream>>>(coord, node_w, x_f, x_b);
    k_build_e<<<(int)(((size_t)ME_ * 64 + 255) / 256), 256, 0, stream>>>(
        evals, tour, best, val_w, emb0, emb1, e_f, e_b);

    // ---- 3 GCN layers
    for (int L = 0; L < 3; ++L) {
        int base = 18 + 12 * L;
        const float* bn_e_g = (const float*)d_in[base + 0];
        const float* bn_e_b = (const float*)d_in[base + 1];
        const float* bn_n_g = (const float*)d_in[base + 2];
        const float* bn_n_b = (const float*)d_in[base + 3];
        k_wmma_gemm64<0, false><<<nblocks(MX_, 1), 256, 0, stream>>>(
            x_b, pk(4 * L + 1), (const float*)d_in[base + 7],  Vx,  nullptr, MX_, nullptr, nullptr, 1);
        k_wmma_gemm64<0, false><<<nblocks(MX_, 1), 256, 0, stream>>>(
            x_b, pk(4 * L + 2), (const float*)d_in[base + 9],  Ux,  nullptr, MX_, nullptr, nullptr, 1);
        k_wmma_gemm64<0, false><<<nblocks(MX_, 1), 256, 0, stream>>>(
            x_b, pk(4 * L + 3), (const float*)d_in[base + 11], Vx2, nullptr, MX_, nullptr, nullptr, 1);
        // fused: e_tmp = e@eU + b + Vx[u] + Vx[v]; gate = sigmoid(e_tmp)
        k_wmma_gemm64<2, false><<<nblocks(ME_, 8), 256, 0, stream>>>(
            e_b, pk(4 * L + 0), (const float*)d_in[base + 5], e_tmp, nullptr, ME_, Vx, gate, 8);
        k_agg<<<(MX_ * 64 + 255) / 256, 256, 0, stream>>>(gate, Vx2, agg, gsum);
        // edge update: e += relu(bn(e_tmp))
        k_stats<<<64, 256, 0, stream>>>(e_tmp, meanb, varb, ME_);
        k_bnres<<<(int)(((size_t)ME_ * 64 + 255) / 256), 256, 0, stream>>>(
            e_f, e_tmp, bn_e_g, bn_e_b, meanb, varb, e_b, (size_t)ME_ * 64);
        // node update: x += relu(bn(Ux + agg/gsum))
        k_xtmp<<<(MX_ * 64 + 255) / 256, 256, 0, stream>>>(Ux, agg, gsum, xtmp);
        k_stats<<<64, 256, 0, stream>>>(xtmp, meanb, varb, MX_);
        k_bnres<<<(MX_ * 64 + 255) / 256, 256, 0, stream>>>(
            x_f, xtmp, bn_n_g, bn_n_b, meanb, varb, x_b, (size_t)MX_ * 64);
    }

    // ---- quad embeddings
    k_edges<<<1, 32, 0, stream>>>(tour, elist, rs, cs);
    k_quad<<<MQ_, 256, 0, stream>>>(e_f, elist, rs, cs, dirb, meta, quad);

    // ---- pre MLP: relu(quad@W1+b1) @ W2 + b2   (K=256 via LDS-staged weights)
    k_wmma_gemm_lds<<<nblocks(MQ_, 4), 256, 256 * 64 * sizeof(bf16), stream>>>(
        quad, pk(12), (const float*)d_in[56], bufA_f, bufA_b, MQ_, 256, 4);
    k_wmma_gemm64<0, true><<<nblocks(MQ_, 8), 256, 0, stream>>>(
        bufA_b, pk(13), (const float*)d_in[58], bufB_f, bufB_b, MQ_, nullptr, nullptr, 8);
    // ---- act MLP: 4 hidden relu layers
    k_wmma_gemm64<1, true><<<nblocks(MQ_, 8), 256, 0, stream>>>(
        bufB_b, pk(14), (const float*)d_in[7],  bufA_f, bufA_b, MQ_, nullptr, nullptr, 8);
    k_wmma_gemm64<1, true><<<nblocks(MQ_, 8), 256, 0, stream>>>(
        bufA_b, pk(15), (const float*)d_in[9],  bufB_f, bufB_b, MQ_, nullptr, nullptr, 8);
    k_wmma_gemm64<1, true><<<nblocks(MQ_, 8), 256, 0, stream>>>(
        bufB_b, pk(16), (const float*)d_in[11], bufA_f, bufA_b, MQ_, nullptr, nullptr, 8);
    k_wmma_gemm64<1, true><<<nblocks(MQ_, 8), 256, 0, stream>>>(
        bufA_b, pk(17), (const float*)d_in[13], bufB_f, bufB_b, MQ_, nullptr, nullptr, 8);
    // ---- out projection 64 -> 1
    k_dotout<<<(MQ_ + 255) / 256, 256, 0, stream>>>(
        bufB_f, (const float*)d_in[14], (const float*)d_in[15], logits, MQ_);

    // ---- action selection + log-prob + output assembly
    k_sample<<<B_, 256, 0, stream>>>(logits, meta, (float*)d_out);
}
